// DeepseekV2Attention_8048768713517
// MI455X (gfx1250) — compile-verified, matrix operand order checked by reading, not run on hardware
//
#include <hip/hip_runtime.h>
#include <math.h>

// ---------------- types ----------------
typedef __bf16 bf16;
typedef __bf16 v16bf __attribute__((ext_vector_type(16)));
typedef float  v8f   __attribute__((ext_vector_type(8)));
typedef unsigned int u32x4 __attribute__((ext_vector_type(4)));
typedef int si32x4 __attribute__((ext_vector_type(4)));
typedef int si32x8 __attribute__((ext_vector_type(8)));

union FragBF { v16bf v; u32x4 u[2]; };

// dims
#define TT   2048
#define HID  2048
#define NH   16
#define QR   1536
#define KVR  512
#define DQK  192
#define DNOPE 128
#define DV   128

// ---------------- CDNA5 async global->LDS copy (ASYNCcnt-tracked) ----------------
__device__ inline void async_ld_b128(bf16* lds_dst, const bf16* gsrc) {
  asm volatile("global_load_async_to_lds_b128 %0, %1, off"
               :: "v"((unsigned)(uintptr_t)lds_dst), "v"(gsrc)
               : "memory");
}
__device__ inline void wait_async() {
  asm volatile("s_wait_asynccnt 0x0" ::: "memory");
}

// ---------------- CDNA5 Tensor Data Mover: 2-D bf16 tile, global -> LDS ----------
// D# group0: count=1 | lds_addr | global_addr(57b) | type=2
// D# group1: data_size=2B; tensor_dim0=w, tensor_dim1=h; tile 32x128; stride=w
// Issued by one wave (EXEC ignored); tracked by TENSORcnt.
__device__ inline void tdm_load_2d_bf16(unsigned lds_addr, const void* gaddr,
                                        unsigned tensor_w, unsigned tensor_h,
                                        unsigned tile_w, unsigned tile_h,
                                        unsigned stride_elems) {
  unsigned long long ga = (unsigned long long)(uintptr_t)gaddr;
  si32x4 g0;
  g0[0] = 1;                                                   // count=1, no gather
  g0[1] = (int)lds_addr;                                       // lds_addr
  g0[2] = (int)(unsigned)(ga & 0xFFFFFFFFull);                 // global_addr[31:0]
  g0[3] = (int)((unsigned)((ga >> 32) & 0x1FFFFFFull) | (2u << 30)); // addr hi | type=2
  si32x8 g1;
  g1[0] = (int)(1u << 16);                                     // data_size=1 (2 bytes)
  g1[1] = (int)((tensor_w & 0xFFFFu) << 16);                   // tensor_dim0 lo
  g1[2] = (int)((tensor_w >> 16) | ((tensor_h & 0xFFFFu) << 16)); // dim0 hi | dim1 lo
  g1[3] = (int)((tensor_h >> 16) | (tile_w << 16));            // dim1 hi | tile_dim0
  g1[4] = (int)tile_h;                                         // tile_dim1 (tile_dim2=0)
  g1[5] = (int)stride_elems;                                   // tensor_dim0_stride lo
  g1[6] = 0;                                                   // stride hi | dim1_stride lo
  g1[7] = 0;
  asm volatile("tensor_load_to_lds %0, %1" :: "s"(g0), "s"(g1) : "memory");
}

// A/B fragment for v_wmma_*_16x16x32 bf16: lane L -> row/col = L&15,
// elements 0..7 -> K = base+0..7, elements 8..15 -> K = base+16..23,
// base = 8 for lanes 16..31. Two 16B LDS loads per lane.
__device__ inline v16bf load_frag16(const bf16* __restrict__ base, int ld_elems) {
  int lane = threadIdx.x & 31;
  int r = lane & 15;
  int kb = (lane & 16) ? 8 : 0;
  const bf16* p = base + (size_t)r * ld_elems;
  FragBF f;
  f.u[0] = *(const u32x4*)(p + kb);
  f.u[1] = *(const u32x4*)(p + kb + 16);
  return f.v;
}

__device__ inline v8f wmma_bf16(v16bf a, v16bf b, v8f c) {
  return __builtin_amdgcn_wmma_f32_16x16x32_bf16(false, a, false, b, (short)0, c,
                                                 false, false);
}

// ---------------- f32 -> bf16 convert ----------------
__global__ __launch_bounds__(256) void cvt_f32_bf16(const float* __restrict__ x,
                                                    bf16* __restrict__ y, size_t n) {
  size_t i = (size_t)blockIdx.x * 256 + threadIdx.x;
  size_t stride = (size_t)gridDim.x * 256;
  for (; i < n; i += stride) y[i] = (bf16)x[i];
}

// ---------------- generic bf16 GEMM: C[M,N] = A[M,K] @ B[K,N] ----------------
// A,B row-major bf16; C f32. M%128==0, N%BNt==0, K%32==0.
// A tile staged by the Tensor Data Mover (one DMA per K-step, wave 0);
// B tile staged transposed through registers. 8 waves: 4 over M x 2 over N.
#define BM 128
#define BK 32
template <int BNt>
__global__ __launch_bounds__(256) void gemm_bf16(const bf16* __restrict__ A,
                                                 const bf16* __restrict__ B,
                                                 float* __restrict__ C,
                                                 int M, int N, int K) {
  constexpr int NTJ = BNt / 32;                // n-tiles per wave
  __shared__ __align__(16) bf16 sA[BM][BK];    // 8 KB
  __shared__ __align__(16) bf16 sBt[BNt][BK];  // 4/8 KB, B tile transposed
  int bm = blockIdx.y * BM;
  int bn = blockIdx.x * BNt;
  int tid = threadIdx.x;
  int wave = tid >> 5;
  int lane = tid & 31;
  int wm = (wave & 3) * 32;            // 4 waves over M
  int wn = (wave >> 2) * (BNt / 2);    // 2 waves over N

  v8f acc[2][NTJ];
#pragma unroll
  for (int i = 0; i < 2; ++i)
#pragma unroll
    for (int j = 0; j < NTJ; ++j) acc[i][j] = (v8f){};

  for (int k0 = 0; k0 < K; k0 += BK) {
    // A 128x32 tile: one TDM descriptor, issued by wave 0 (EXEC-independent DMA)
    if (tid < 32) {
      tdm_load_2d_bf16((unsigned)(uintptr_t)&sA[0][0],
                       &A[(size_t)bm * K + k0],
                       (unsigned)K, (unsigned)M, BK, BM, (unsigned)K);
    }
    // B tile transposed: rows k0..k0+31, cols bn..bn+BNt-1 -> sBt[n][k]
    {
      int kk = tid >> 3;                 // 0..31
      int c0 = (tid & 7) * (BNt / 8);    // chunk of BNt/8 cols
#pragma unroll
      for (int l = 0; l < BNt / 64; ++l) {
        int c = c0 + l * 8;
        bf16 tmp[8];
        *(u32x4*)tmp = *(const u32x4*)&B[(size_t)(k0 + kk) * N + bn + c];
#pragma unroll
        for (int j = 0; j < 8; ++j) sBt[c + j][kk] = tmp[j];
      }
      // prefetch next K-step's B rows into cache (global_prefetch_b8)
      if (k0 + BK < K)
        __builtin_prefetch(&B[(size_t)(k0 + BK + kk) * N + bn + c0], 0, 1);
    }
    if (tid < 32) __builtin_amdgcn_s_wait_tensorcnt(0);
    __syncthreads();
    v16bf bfrag[NTJ];
#pragma unroll
    for (int j = 0; j < NTJ; ++j) bfrag[j] = load_frag16(&sBt[wn + j * 16][0], BK);
#pragma unroll
    for (int i = 0; i < 2; ++i) {
      v16bf afrag = load_frag16(&sA[wm + i * 16][0], BK);
#pragma unroll
      for (int j = 0; j < NTJ; ++j) acc[i][j] = wmma_bf16(afrag, bfrag[j], acc[i][j]);
    }
    __syncthreads();
  }
  // C fragment layout: lane holds col = lane&15; VGPR r -> row r (+8 for lanes>=16)
  int col = lane & 15;
  int rhalf = (lane & 16) ? 8 : 0;
#pragma unroll
  for (int i = 0; i < 2; ++i)
#pragma unroll
    for (int j = 0; j < NTJ; ++j) {
      float* cp = C + (size_t)(bm + wm + i * 16 + rhalf) * N + bn + wn + j * 16 + col;
#pragma unroll
      for (int r = 0; r < 8; ++r) cp[(size_t)r * N] = acc[i][j][r];
    }
}

// ---------------- RMS norm -> bf16 ----------------
__global__ __launch_bounds__(256) void rmsnorm_bf16(const float* __restrict__ x,
                                                    const float* __restrict__ w,
                                                    bf16* __restrict__ y,
                                                    int R, int stride) {
  int t = blockIdx.x;
  const float* row = x + (size_t)t * stride;
  float s = 0.f;
  for (int i = threadIdx.x; i < R; i += 256) { float v = row[i]; s += v * v; }
#pragma unroll
  for (int m = 16; m >= 1; m >>= 1) s += __shfl_xor(s, m, 32);
  __shared__ float red[8];
  int wave = threadIdx.x >> 5, lane = threadIdx.x & 31;
  if (lane == 0) red[wave] = s;
  __syncthreads();
  float tot = 0.f;
#pragma unroll
  for (int i = 0; i < 8; ++i) tot += red[i];
  float inv = rsqrtf(tot / (float)R + 1e-6f);
  for (int i = threadIdx.x; i < R; i += 256)
    y[(size_t)t * R + i] = (bf16)(row[i] * inv * w[i]);
}

// ---------------- YaRN RoPE + head-major repack ----------------
// Qa/Ka: [H][T][192] bf16; Va: [H][T][128] bf16
__global__ __launch_bounds__(256) void rope_pack(const int* __restrict__ positions,
                                                 const float* __restrict__ qfull,   // [T][H*192]
                                                 const float* __restrict__ kvfull,  // [T][H*256]
                                                 const float* __restrict__ latent,  // [T][576]
                                                 bf16* __restrict__ Qa,
                                                 bf16* __restrict__ Ka,
                                                 bf16* __restrict__ Va) {
  int t = blockIdx.x;
  int tid = threadIdx.x;
  __shared__ float cs[32], sn[32];
  if (tid < 32) {
    float fi = (float)tid;
    float pf = __powf(10000.f, (2.f * fi) / 64.f);
    float inv_extra = 1.f / pf;
    float inv_inter = 1.f / (40.f * pf);
    // yarn ramp: low=floor(corr(32))=10, high=min(ceil(corr(1)),31)=23
    float ramp = fminf(fmaxf((fi - 10.f) / 13.f, 0.f), 1.f);
    float mask = 1.f - ramp;
    float invf = inv_inter * (1.f - mask) + inv_extra * mask;
    float ang = (float)positions[t] * invf;
    cs[tid] = __cosf(ang);
    sn[tid] = __sinf(ang);
  }
  __syncthreads();
  // Q
  for (int idx = tid; idx < NH * DQK; idx += 256) {
    int hh = idx / DQK, d = idx % DQK;
    const float* qrow = qfull + (size_t)t * (NH * DQK) + hh * DQK;
    float val;
    if (d < DNOPE) {
      val = qrow[d];
    } else {
      int j = d - DNOPE, i2 = j >> 1;
      float x1 = qrow[DNOPE + 2 * i2], x2 = qrow[DNOPE + 2 * i2 + 1];
      val = (j & 1) ? (x2 * cs[i2] + x1 * sn[i2]) : (x1 * cs[i2] - x2 * sn[i2]);
    }
    Qa[((size_t)hh * TT + t) * DQK + d] = (bf16)val;
  }
  // K (k_pe broadcast across heads)
  for (int idx = tid; idx < NH * DQK; idx += 256) {
    int hh = idx / DQK, d = idx % DQK;
    float val;
    if (d < DNOPE) {
      val = kvfull[(size_t)t * (NH * 256) + hh * 256 + d];
    } else {
      int j = d - DNOPE, i2 = j >> 1;
      const float* kp = latent + (size_t)t * 576 + KVR;
      float x1 = kp[2 * i2], x2 = kp[2 * i2 + 1];
      val = (j & 1) ? (x2 * cs[i2] + x1 * sn[i2]) : (x1 * cs[i2] - x2 * sn[i2]);
    }
    Ka[((size_t)hh * TT + t) * DQK + d] = (bf16)val;
  }
  // V
  for (int idx = tid; idx < NH * DV; idx += 256) {
    int hh = idx / DV, d = idx % DV;
    Va[((size_t)hh * TT + t) * DV + d] =
        (bf16)kvfull[(size_t)t * (NH * 256) + hh * 256 + DNOPE + d];
  }
}

// ---------------- flash attention (causal, online softmax) ----------------
// grid: (T/128, H), block 256 (8 waves); each wave owns 16 query rows.
__global__ __launch_bounds__(256) void flash_attn(const bf16* __restrict__ Q,  // [H][T][192]
                                                  const bf16* __restrict__ Kh, // [H][T][192]
                                                  const bf16* __restrict__ Vh, // [H][T][128]
                                                  bf16* __restrict__ O,        // [T][H*128]
                                                  float scaling) {
  __shared__ __align__(16) bf16 sK[32][DQK];     // 12 KB, key-major
  __shared__ __align__(16) bf16 sVt[DV][32];     // 8 KB, vdim-major
  __shared__ __align__(16) bf16 sP[8][16][32];   // per-wave P tile, 8 KB
  int h = blockIdx.y;
  int qb = blockIdx.x * 128;
  int tid = threadIdx.x, lane = tid & 31, wave = tid >> 5;
  int qrow0 = qb + wave * 16;
  int col = lane & 15;
  int rhalf = (lane & 16) ? 8 : 0;

  // load Q fragments (6 x 16x32 over d=192) straight from global
  v16bf qf[6];
  {
    int r = lane & 15;
    int kb = (lane & 16) ? 8 : 0;
    const bf16* p = Q + ((size_t)h * TT + qrow0 + r) * DQK;
#pragma unroll
    for (int f = 0; f < 6; ++f) {
      FragBF fr;
      fr.u[0] = *(const u32x4*)(p + f * 32 + kb);
      fr.u[1] = *(const u32x4*)(p + f * 32 + kb + 16);
      qf[f] = fr.v;
    }
  }

  v8f acc[8];
#pragma unroll
  for (int j = 0; j < 8; ++j) acc[j] = (v8f){};
  float rowmax[8], rowsum[8];
#pragma unroll
  for (int r = 0; r < 8; ++r) { rowmax[r] = -__builtin_inff(); rowsum[r] = 0.f; }

  int nblocks = (qb + 128) >> 5;  // causal: keys up to end of this q block
  for (int b = 0; b < nblocks; ++b) {
    int k0 = b * 32;
    __syncthreads();
    // stage K block [32][192] via async global->LDS
    {
      int kk = tid >> 3;          // 0..31
      int c0 = (tid & 7) * 8;     // 0..56
      const bf16* src = Kh + ((size_t)h * TT + k0 + kk) * DQK;
#pragma unroll
      for (int it = 0; it < 3; ++it) {
        int c = c0 + it * 64;
        async_ld_b128(&sK[kk][c], src + c);
      }
    }
    // stage V block transposed [128][32]
    {
      int kk = tid >> 3;
      int d0 = (tid & 7) * 16;
      const bf16* src = Vh + ((size_t)h * TT + k0 + kk) * DV + d0;
      bf16 tmp[16];
      *(u32x4*)&tmp[0] = *(const u32x4*)src;
      *(u32x4*)&tmp[8] = *(const u32x4*)(src + 8);
#pragma unroll
      for (int j = 0; j < 16; ++j) sVt[d0 + j][kk] = tmp[j];
    }
    wait_async();
    __syncthreads();

    // S = Q K^T for two 16-key sub-tiles
    v8f s0 = (v8f){}, s1 = (v8f){};
#pragma unroll
    for (int f = 0; f < 6; ++f) {
      v16bf b0 = load_frag16(&sK[0][f * 32], DQK);
      v16bf b1 = load_frag16(&sK[16][f * 32], DQK);
      s0 = wmma_bf16(qf[f], b0, s0);
      s1 = wmma_bf16(qf[f], b1, s1);
    }

    // scale + causal mask + online softmax (row r lives in VGPR slot r, halves +/-8)
    float p0[8], p1[8];
#pragma unroll
    for (int r = 0; r < 8; ++r) {
      int qrow = qrow0 + rhalf + r;
      float v0 = s0[r] * scaling;
      float v1 = s1[r] * scaling;
      if (k0 + col > qrow)      v0 = -__builtin_inff();
      if (k0 + 16 + col > qrow) v1 = -__builtin_inff();
      float m = fmaxf(v0, v1);
      m = fmaxf(m, __shfl_xor(m, 1, 32));
      m = fmaxf(m, __shfl_xor(m, 2, 32));
      m = fmaxf(m, __shfl_xor(m, 4, 32));
      m = fmaxf(m, __shfl_xor(m, 8, 32));
      float mnew = fmaxf(rowmax[r], m);
      float sc = __expf(rowmax[r] - mnew);
      float e0 = __expf(v0 - mnew);
      float e1 = __expf(v1 - mnew);
      float esum = e0 + e1;
      esum += __shfl_xor(esum, 1, 32);
      esum += __shfl_xor(esum, 2, 32);
      esum += __shfl_xor(esum, 4, 32);
      esum += __shfl_xor(esum, 8, 32);
      rowsum[r] = rowsum[r] * sc + esum;
      rowmax[r] = mnew;
      p0[r] = e0; p1[r] = e1;
#pragma unroll
      for (int j = 0; j < 8; ++j) acc[j][r] *= sc;
    }

    // re-layout P (C-frag) -> A-frag via per-wave LDS
#pragma unroll
    for (int r = 0; r < 8; ++r) {
      sP[wave][rhalf + r][col]      = (bf16)p0[r];
      sP[wave][rhalf + r][16 + col] = (bf16)p1[r];
    }
    v16bf pf = load_frag16(&sP[wave][0][0], 32);
#pragma unroll
    for (int j = 0; j < 8; ++j) {
      v16bf vf = load_frag16(&sVt[j * 16][0], 32);
      acc[j] = wmma_bf16(pf, vf, acc[j]);
    }
  }

  // epilogue: normalize, write [T][H*128] bf16 (row-major for final GEMM)
#pragma unroll
  for (int j = 0; j < 8; ++j)
#pragma unroll
    for (int r = 0; r < 8; ++r) {
      int qrow = qrow0 + rhalf + r;
      float o = acc[j][r] / rowsum[r];
      O[(size_t)qrow * (NH * DV) + h * DV + j * 16 + col] = (bf16)o;
    }
}

// ---------------- launch ----------------
extern "C" void kernel_launch(void* const* d_in, const int* in_sizes, int n_in,
                              void* d_out, int out_size, void* d_ws, size_t ws_size,
                              hipStream_t stream) {
  const int*   positions = (const int*)d_in[0];
  const float* hidden    = (const float*)d_in[1];
  const float* wq_a      = (const float*)d_in[2];
  const float* q_norm_w  = (const float*)d_in[3];
  const float* wq_b      = (const float*)d_in[4];
  const float* wkv_a     = (const float*)d_in[5];
  const float* kv_norm_w = (const float*)d_in[6];
  const float* wkv_b     = (const float*)d_in[7];
  const float* wo        = (const float*)d_in[8];
  float* out = (float*)d_out;

  char* p = (char*)d_ws;
  auto alloc = [&](size_t bytes) -> char* {
    char* r = p;
    p += (bytes + 255) & ~(size_t)255;
    return r;
  };
  bf16* hidden_bf = (bf16*)alloc((size_t)TT * HID * 2);
  bf16* wq_a_bf   = (bf16*)alloc((size_t)HID * QR * 2);
  bf16* wq_b_bf   = (bf16*)alloc((size_t)QR * (NH * DQK) * 2);
  bf16* wkv_a_bf  = (bf16*)alloc((size_t)HID * (KVR + 64) * 2);
  bf16* wkv_b_bf  = (bf16*)alloc((size_t)KVR * (NH * 256) * 2);
  bf16* wo_bf     = (bf16*)alloc((size_t)(NH * DV) * HID * 2);
  float* q_lat    = (float*)alloc((size_t)TT * QR * 4);
  bf16*  q_bf     = (bf16*)alloc((size_t)TT * QR * 2);
  float* qfull    = (float*)alloc((size_t)TT * NH * DQK * 4);
  float* latent   = (float*)alloc((size_t)TT * (KVR + 64) * 4);
  bf16*  kv_bf    = (bf16*)alloc((size_t)TT * KVR * 2);
  float* kvfull   = (float*)alloc((size_t)TT * NH * 256 * 4);
  bf16*  Qa       = (bf16*)alloc((size_t)NH * TT * DQK * 2);
  bf16*  Ka       = (bf16*)alloc((size_t)NH * TT * DQK * 2);
  bf16*  Va       = (bf16*)alloc((size_t)NH * TT * DV * 2);
  bf16*  attn_bf  = (bf16*)alloc((size_t)TT * NH * DV * 2);

  double mm = 0.1 * 0.707 * log(40.0) + 1.0;
  float scaling = (float)(pow((double)DQK, -0.5) * mm * mm);

  auto cvt = [&](const float* src, bf16* dst, size_t n) {
    cvt_f32_bf16<<<2048, 256, 0, stream>>>(src, dst, n);
  };
  cvt(hidden, hidden_bf, (size_t)TT * HID);
  cvt(wq_a,  wq_a_bf,  (size_t)HID * QR);
  cvt(wq_b,  wq_b_bf,  (size_t)QR * NH * DQK);
  cvt(wkv_a, wkv_a_bf, (size_t)HID * (KVR + 64));
  cvt(wkv_b, wkv_b_bf, (size_t)KVR * NH * 256);
  cvt(wo,    wo_bf,    (size_t)NH * DV * HID);

  auto gemm128 = [&](const bf16* A, const bf16* B, float* C, int M, int N, int K) {
    dim3 grid(N / 128, M / BM);
    gemm_bf16<128><<<grid, 256, 0, stream>>>(A, B, C, M, N, K);
  };
  auto gemm64 = [&](const bf16* A, const bf16* B, float* C, int M, int N, int K) {
    dim3 grid(N / 64, M / BM);
    gemm_bf16<64><<<grid, 256, 0, stream>>>(A, B, C, M, N, K);
  };

  // q = rms(hidden @ wq_a) ; q @ wq_b
  gemm128(hidden_bf, wq_a_bf, q_lat, TT, QR, HID);
  rmsnorm_bf16<<<TT, 256, 0, stream>>>(q_lat, q_norm_w, q_bf, QR, QR);
  gemm128(q_bf, wq_b_bf, qfull, TT, NH * DQK, QR);

  // latent = hidden @ wkv_a ; kv = rms(latent[:,:512]) @ wkv_b
  gemm64(hidden_bf, wkv_a_bf, latent, TT, KVR + 64, HID);
  rmsnorm_bf16<<<TT, 256, 0, stream>>>(latent, kv_norm_w, kv_bf, KVR, KVR + 64);
  gemm128(kv_bf, wkv_b_bf, kvfull, TT, NH * 256, KVR);

  // RoPE + head-major repack
  rope_pack<<<TT, 256, 0, stream>>>(positions, qfull, kvfull, latent, Qa, Ka, Va);

  // causal flash attention
  flash_attn<<<dim3(TT / 128, NH), 256, 0, stream>>>(Qa, Ka, Va, attn_bf, scaling);

  // out = attn @ wo
  gemm128(attn_bf, wo_bf, out, TT, HID, NH * DV);
}
